// AttentionTransformer_67070209294683
// MI455X (gfx1250) — matrix operand order repaired and checked by reading, not compiled, not run on hardware
//
#include <hip/hip_runtime.h>
#include <hip/hip_bf16.h>

// ---------------------------------------------------------------------------
// TabNet attentive-transformer mask on MI455X (gfx1250, wave32, WMMA).
//   h    = a @ W (+b)            -- b cancels exactly under ghost-BN mean
//   h_bn = ghostBN(h) per 128-row chunk (biased var, eps=1e-5)
//   mask = sparsemax(h_bn * priors) along D=256
// One workgroup (256 thr = 8 waves) per 128-row ghost batch; whole chain
// fused in LDS. HBM-bound (~768MB moved => ~33us floor at 23.3TB/s);
// f16 WMMA (16x16x32, f32 accum) gives ample compute headroom.
// ---------------------------------------------------------------------------

typedef _Float16 v16h __attribute__((ext_vector_type(16)));
typedef float    v8f  __attribute__((ext_vector_type(8)));

union F16x16 { v16h v; uint4 q[2]; };

#define D_DIM   256
#define VBS     128
#define AS      264              // a-chunk LDS row stride (halves), 16B-aligned rows
#define WS      264              // W^T  LDS row stride (halves)
#define HS      260              // h    LDS row stride (floats), 16B-aligned rows
#define A_BYTES (VBS * AS * 2)   // 67584
#define W_BYTES (D_DIM * WS * 2) // 135168  (h reuse: 128*260*4 = 133120 <= this)
#define SMEM_BYTES (A_BYTES + W_BYTES)

__device__ __forceinline__ float wred_sum(float v) {
  v += __shfl_xor(v, 16, 32); v += __shfl_xor(v, 8, 32);
  v += __shfl_xor(v,  4, 32); v += __shfl_xor(v, 2, 32);
  v += __shfl_xor(v,  1, 32);
  return v;
}
__device__ __forceinline__ float wred_max(float v) {
  v = fmaxf(v, __shfl_xor(v, 16, 32)); v = fmaxf(v, __shfl_xor(v, 8, 32));
  v = fmaxf(v, __shfl_xor(v,  4, 32)); v = fmaxf(v, __shfl_xor(v, 2, 32));
  v = fmaxf(v, __shfl_xor(v,  1, 32));
  return v;
}

// Prologue: W[k][n] f32 -> Wt[n][k] f16 (compact 256x256) in workspace.
__global__ __launch_bounds__(256) void wt_transpose_kernel(
    const float* __restrict__ W, _Float16* __restrict__ wt) {
  int g  = blockIdx.x * 256 + threadIdx.x;   // 16384 threads, 1 float4 each
  int k  = g >> 6;
  int n0 = (g & 63) * 4;
  float4 w = *(const float4*)(W + (size_t)k * D_DIM + n0);
  wt[(size_t)(n0 + 0) * D_DIM + k] = (_Float16)w.x;
  wt[(size_t)(n0 + 1) * D_DIM + k] = (_Float16)w.y;
  wt[(size_t)(n0 + 2) * D_DIM + k] = (_Float16)w.z;
  wt[(size_t)(n0 + 3) * D_DIM + k] = (_Float16)w.w;
}

__global__ __launch_bounds__(256) void tabnet_mask_kernel(
    const float* __restrict__ a, const float* __restrict__ priors,
    const _Float16* __restrict__ wt, const float* __restrict__ gamma,
    const float* __restrict__ beta, float* __restrict__ out) {
  extern __shared__ char smem[];
  _Float16* aS = (_Float16*)smem;
  _Float16* wS = (_Float16*)(smem + A_BYTES);
  float*    hS = (float*)(smem + A_BYTES);   // reuses W^T region after GEMM

  const int t    = threadIdx.x;
  const int lane = t & 31;
  const int wave = t >> 5;        // 0..7, owns columns [wave*32, wave*32+32)
  const int m    = lane & 15;
  const int hi   = lane >> 4;
  const int rowbase = blockIdx.x * VBS;

  // ---- Stage 1: a-chunk f32->f16 into LDS; W^T f16 copy into LDS ----------
  {
    int col4 = (t & 63) * 4;
    int r0   = t >> 6;
    #pragma unroll 4
    for (int j = 0; j < 32; ++j) {
      int row = r0 + j * 4;
      float4 av = *(const float4*)(a + (size_t)(rowbase + row) * D_DIM + col4);
      union { _Float16 h[4]; unsigned long long u; } pk;
      pk.h[0] = (_Float16)av.x; pk.h[1] = (_Float16)av.y;
      pk.h[2] = (_Float16)av.z; pk.h[3] = (_Float16)av.w;
      *(unsigned long long*)(aS + (size_t)row * AS + col4) = pk.u;
    }
    #pragma unroll 4
    for (int j = 0; j < 32; ++j) {
      int i = t + 256 * j;        // 8192 16-byte chunks of compact Wt
      int n = i >> 5;
      int k = (i & 31) * 8;
      uint4 c = *(const uint4*)(wt + (size_t)n * D_DIM + k);
      *(uint4*)(wS + (size_t)n * WS + k) = c;
    }
  }
  __syncthreads();

  // ---- Stage 2: GEMM, 8 M-tiles x 2 N-tiles per wave, K in steps of 32 ----
  v8f acc[16];
  #pragma unroll
  for (int i = 0; i < 16; ++i) acc[i] = (v8f)0.0f;

  for (int ks = 0; ks < 8; ++ks) {
    // B fragments (contiguous k order, same as A): lane n -> col n&15, K = (n>>4)*16 + 0..15
    F16x16 b0, b1;
    {
      const _Float16* p0 = wS + (size_t)(wave * 32 + m) * WS + ks * 32 + hi * 16;
      const _Float16* p1 = p0 + (size_t)16 * WS;
      b0.q[0] = *(const uint4*)(p0); b0.q[1] = *(const uint4*)(p0 + 8);
      b1.q[0] = *(const uint4*)(p1); b1.q[1] = *(const uint4*)(p1 + 8);
    }
    #pragma unroll
    for (int mt = 0; mt < 8; ++mt) {
      // A fragment: lane -> row m, halves [hi*16, hi*16+16)
      F16x16 af;
      const _Float16* pa = aS + (size_t)(mt * 16 + m) * AS + ks * 32 + hi * 16;
      af.q[0] = *(const uint4*)(pa);
      af.q[1] = *(const uint4*)(pa + 8);
      acc[mt * 2 + 0] = __builtin_amdgcn_wmma_f32_16x16x32_f16(
          false, af.v, false, b0.v, (short)0, acc[mt * 2 + 0], false, false);
      acc[mt * 2 + 1] = __builtin_amdgcn_wmma_f32_16x16x32_f16(
          false, af.v, false, b1.v, (short)0, acc[mt * 2 + 1], false, false);
    }
  }
  __syncthreads();   // all waves done reading wS before it becomes hS

  // ---- Stage 3: spill accumulators to LDS h (C layout: M=r+hi*8, N=m) -----
  #pragma unroll
  for (int mt = 0; mt < 8; ++mt)
    #pragma unroll
    for (int nt = 0; nt < 2; ++nt) {
      int col = wave * 32 + nt * 16 + m;
      #pragma unroll
      for (int r = 0; r < 8; ++r) {
        int row = mt * 16 + hi * 8 + r;
        hS[(size_t)row * HS + col] = acc[mt * 2 + nt][r];
      }
    }
  __syncthreads();

  // ---- Stage 4: ghost BN (col stats over 128 rows) + multiply priors ------
  {
    int c = t;                      // one thread per feature column
    float s = 0.f, s2 = 0.f;
    for (int r = 0; r < VBS; ++r) {
      float x = hS[(size_t)r * HS + c];
      s += x; s2 += x * x;
    }
    float mean = s * (1.0f / VBS);
    float var  = s2 * (1.0f / VBS) - mean * mean;
    float inv  = rsqrtf(var + 1e-5f) * gamma[c];
    float be   = beta[c];
    size_t pb  = (size_t)rowbase * D_DIM + c;
    for (int r = 0; r < VBS; ++r) {
      float x  = hS[(size_t)r * HS + c];
      float xn = (x - mean) * inv + be;
      hS[(size_t)r * HS + c] = xn * priors[pb + (size_t)r * D_DIM];
    }
  }
  __syncthreads();

  // ---- Stage 5: sparsemax per row (Michelot exact threshold, wave32) ------
  for (int rr = 0; rr < 16; ++rr) {
    int row = wave * 16 + rr;
    const float* hp = hS + (size_t)row * HS + lane * 8;
    float4 u0 = *(const float4*)(hp);
    float4 u1 = *(const float4*)(hp + 4);
    float z[8] = {u0.x, u0.y, u0.z, u0.w, u1.x, u1.y, u1.z, u1.w};

    float mx = z[0];
    #pragma unroll
    for (int i = 1; i < 8; ++i) mx = fmaxf(mx, z[i]);
    mx = wred_max(mx);

    float sm = 0.f;
    #pragma unroll
    for (int i = 0; i < 8; ++i) { z[i] -= mx; sm += z[i]; }
    sm = wred_sum(sm);

    float tau = (sm - 1.0f) * (1.0f / 256.0f);   // full-support start
    for (int it = 0; it < 256; ++it) {           // support shrinks each step
      float s = 0.f, c = 0.f;
      #pragma unroll
      for (int i = 0; i < 8; ++i)
        if (z[i] > tau) { s += z[i]; c += 1.0f; }
      s = wred_sum(s); c = wred_sum(c);
      float ntau = (s - 1.0f) / c;               // wave-uniform
      if (ntau == tau) break;
      tau = ntau;
    }

    float* op = out + ((size_t)(rowbase + row) * D_DIM + lane * 8);
    float4 o0, o1;
    o0.x = fmaxf(z[0] - tau, 0.f); o0.y = fmaxf(z[1] - tau, 0.f);
    o0.z = fmaxf(z[2] - tau, 0.f); o0.w = fmaxf(z[3] - tau, 0.f);
    o1.x = fmaxf(z[4] - tau, 0.f); o1.y = fmaxf(z[5] - tau, 0.f);
    o1.z = fmaxf(z[6] - tau, 0.f); o1.w = fmaxf(z[7] - tau, 0.f);
    *(float4*)(op)     = o0;
    *(float4*)(op + 4) = o1;
  }
}

extern "C" void kernel_launch(void* const* d_in, const int* in_sizes, int n_in,
                              void* d_out, int out_size, void* d_ws, size_t ws_size,
                              hipStream_t stream) {
  const float* a      = (const float*)d_in[0];
  const float* priors = (const float*)d_in[1];
  const float* W      = (const float*)d_in[2];
  // d_in[3] = bias b: a per-column constant, cancelled exactly by ghost-BN
  const float* gamma  = (const float*)d_in[4];
  const float* beta   = (const float*)d_in[5];
  float* out = (float*)d_out;
  _Float16* wt = (_Float16*)d_ws;              // 256*256*2 = 128KB of d_ws

  int B = in_sizes[0] / D_DIM;
  int chunks = B / VBS;

  wt_transpose_kernel<<<64, 256, 0, stream>>>(W, wt);
  tabnet_mask_kernel<<<chunks, 256, SMEM_BYTES, stream>>>(
      a, priors, wt, gamma, beta, out);
}
